// MoELayer_75703093559469
// MI455X (gfx1250) — matrix operand order repaired and checked by reading, not compiled, not run on hardware
//
#include <hip/hip_runtime.h>

// MoE: B=4,T=4096,D=1024,F=2048,E=8,K=2 -> N=16384 tokens, NK=32768 assignments.
// Compute-bound (~1000 flop/byte): bf16 WMMA for both expert GEMMs.
// Pipeline: convert x->bf16, transpose W1/W2->bf16 (one pass), route (f32),
// grouped GEMM1 (silu) -> H bf16, grouped GEMM2 -> atomic f32 accumulate into y.
// GEMM inner loop: async global->LDS tile copies (ASYNCcnt) double-buffered,
// 16 v_wmma_f32_16x16x32_bf16 per wave between barriers (KC=64).

#define Ntok 16384
#define Dm   1024
#define Fm   2048
#define Em   8
#define NKa  (Ntok * 2)
#define BM   128
#define BN   128
#define KC   64
#define PA   72   // LDS row pitch in elems; 144B rows keep b128 alignment, spread banks
#define PAT  40   // transpose-kernel tile pitch

typedef __attribute__((ext_vector_type(16))) __bf16 v16bf;
typedef __attribute__((ext_vector_type(8)))  __bf16 v8bf;
typedef __attribute__((ext_vector_type(8)))  float  v8f;
typedef __attribute__((ext_vector_type(8)))  unsigned short v8us;
typedef int v4i_vs __attribute__((vector_size(16)));

union V16U { v16bf v; v8bf h[2]; };

#if __has_builtin(__builtin_amdgcn_global_load_async_to_lds_b128)
#define USE_ASYNC_LDS 1
#else
#define USE_ASYNC_LDS 0
#endif

__device__ __forceinline__ void async_cp16(const void* g, void* l) {
#if USE_ASYNC_LDS
  __builtin_amdgcn_global_load_async_to_lds_b128(
      (__attribute__((address_space(1))) v4i_vs*)(g),
      (__attribute__((address_space(3))) v4i_vs*)(l), 0, 0);
#else
  (void)g; (void)l;
#endif
}

__device__ __forceinline__ void wait_async0() {
#if __has_builtin(__builtin_amdgcn_s_wait_asynccnt)
  __builtin_amdgcn_s_wait_asynccnt(0);
#else
  asm volatile("s_wait_asynccnt 0x0" ::: "memory");
#endif
}

__device__ __forceinline__ unsigned short f2bfu(float f) {
  unsigned u = __builtin_bit_cast(unsigned, f);
  return (unsigned short)((u + 0x7FFFu + ((u >> 16) & 1u)) >> 16);
}
__device__ __forceinline__ float siluf(float a) {
  return a * __builtin_amdgcn_rcpf(1.0f + __expf(-a));
}

// ---------------- zero / meta ----------------
__global__ void zero_meta_kernel(int* cnt, float* ssum) {
  int i = threadIdx.x;
  if (i < Em) { cnt[i] = 0; ssum[i] = 0.0f; }
}

__global__ void zero_y_kernel(float* __restrict__ y) {
  size_t i = (size_t)blockIdx.x * blockDim.x + threadIdx.x;
  if (i < (size_t)Ntok * Dm / 4) ((float4*)y)[i] = make_float4(0.f, 0.f, 0.f, 0.f);
}

// ---------------- x -> bf16 ----------------
__global__ __launch_bounds__(256) void cvt_x_kernel(const float* __restrict__ x,
                                                    unsigned short* __restrict__ xb) {
  const size_t i = ((size_t)blockIdx.x * 256 + threadIdx.x) * 8;
  float4 a = *(const float4*)(x + i);
  float4 b = *(const float4*)(x + i + 4);
  v8us v;
  v[0]=f2bfu(a.x); v[1]=f2bfu(a.y); v[2]=f2bfu(a.z); v[3]=f2bfu(a.w);
  v[4]=f2bfu(b.x); v[5]=f2bfu(b.y); v[6]=f2bfu(b.z); v[7]=f2bfu(b.w);
  *(v8us*)(xb + i) = v;
}

// ------------- per-expert transpose f32[R][C] -> bf16[C][R] -------------
__global__ __launch_bounds__(256) void transpose_bf16_kernel(
    const float* __restrict__ src, unsigned short* __restrict__ dst, int R, int C) {
  const int e = blockIdx.z;
  src += (size_t)e * R * C;
  dst += (size_t)e * R * C;
  const int c0 = blockIdx.x * 64;
  const int r0 = blockIdx.y * 32;
  __shared__ __align__(16) unsigned short T[64][PAT];
  const int tid = threadIdx.x;
  {
    const int rl = tid >> 3, cl = (tid & 7) * 8;   // 32 rows x 64 cols, coalesced read
    const float* sp = src + (size_t)(r0 + rl) * C + c0 + cl;
    float4 a = *(const float4*)sp;
    float4 b = *(const float4*)(sp + 4);
    T[cl+0][rl]=f2bfu(a.x); T[cl+1][rl]=f2bfu(a.y); T[cl+2][rl]=f2bfu(a.z); T[cl+3][rl]=f2bfu(a.w);
    T[cl+4][rl]=f2bfu(b.x); T[cl+5][rl]=f2bfu(b.y); T[cl+6][rl]=f2bfu(b.z); T[cl+7][rl]=f2bfu(b.w);
  }
  __syncthreads();
  {
    const int cl = tid >> 2, rl = (tid & 3) * 8;   // 64 rows x 32 cols, 16B writes
    v8us v = *(const v8us*)&T[cl][rl];
    *(v8us*)(dst + (size_t)(c0 + cl) * R + r0 + rl) = v;
  }
}

// ---------------- routing: logits, top-2, softmax ----------------
__global__ __launch_bounds__(256) void route_kernel(
    const float* __restrict__ x, const float* __restrict__ Wg,
    int* __restrict__ cnt, float* __restrict__ ssum,
    int* __restrict__ te, float* __restrict__ tw) {
  const int wid  = threadIdx.x >> 5;
  const int lane = threadIdx.x & 31;
  const int t = blockIdx.x * 8 + wid;
  const float* xr = x + (size_t)t * Dm;
  float acc[Em];
#pragma unroll
  for (int e = 0; e < Em; ++e) acc[e] = 0.0f;
  for (int j = 0; j < Dm / 32; ++j) {
    const int d = j * 32 + lane;
    const float xv = xr[d];
    const float* wg = Wg + (size_t)d * Em;
    float4 g0 = *(const float4*)wg;
    float4 g1 = *(const float4*)(wg + 4);
    acc[0] += xv * g0.x; acc[1] += xv * g0.y; acc[2] += xv * g0.z; acc[3] += xv * g0.w;
    acc[4] += xv * g1.x; acc[5] += xv * g1.y; acc[6] += xv * g1.z; acc[7] += xv * g1.w;
  }
#pragma unroll
  for (int off = 16; off >= 1; off >>= 1) {
#pragma unroll
    for (int e = 0; e < Em; ++e) acc[e] += __shfl_xor(acc[e], off, 32);
  }
  if (lane == 0) {
    int i0 = 0; float s0 = acc[0];
#pragma unroll
    for (int e = 1; e < Em; ++e) if (acc[e] > s0) { s0 = acc[e]; i0 = e; }
    int i1 = -1; float s1 = -3.4e38f;
#pragma unroll
    for (int e = 0; e < Em; ++e) if (e != i0 && acc[e] > s1) { s1 = acc[e]; i1 = e; }
    const float ex = __expf(s1 - s0);
    const float w0 = 1.0f / (1.0f + ex);
    const float w1 = ex * w0;
    te[2 * t + 0] = i0; te[2 * t + 1] = i1;
    tw[2 * t + 0] = w0; tw[2 * t + 1] = w1;
    atomicAdd(&cnt[i0], 1); atomicAdd(&cnt[i1], 1);
    atomicAdd(&ssum[i0], w0); atomicAdd(&ssum[i1], w1);
  }
}

// ---------------- prefix + aux loss ----------------
__global__ void prefix_aux_kernel(const int* cnt, const float* ssum,
                                  int* baseo, int* curs, float* y) {
  int b = 0;
  for (int e = 0; e < Em; ++e) { baseo[e] = b; curs[e] = b; b += cnt[e]; }
  baseo[Em] = b;
  float tot = 0.0f;
  for (int e = 0; e < Em; ++e) tot += ssum[e];
  float aux = 0.0f;
  for (int e = 0; e < Em; ++e) {
    const float l = ssum[e] / tot;
    aux += l * __logf(l + 1e-9f);
  }
  y[(size_t)Ntok * Dm] = aux;
}

// ---------------- scatter tokens into expert segments ----------------
__global__ __launch_bounds__(256) void scatter_kernel(
    const int* __restrict__ te, const float* __restrict__ tw,
    int* __restrict__ curs, int* __restrict__ rowmap, float* __restrict__ wrow) {
  const int t = blockIdx.x * 256 + threadIdx.x;
  if (t >= Ntok) return;
#pragma unroll
  for (int k = 0; k < 2; ++k) {
    const int e = te[2 * t + k];
    const int pos = atomicAdd(&curs[e], 1);
    rowmap[pos] = t;
    wrow[pos] = tw[2 * t + k];
  }
}

// ================== shared GEMM inner-loop machinery ==================
#define GEMM_PREAMBLE()                                                       \
  const int tid = threadIdx.x;                                                \
  const int lane = tid & 31, wid = tid >> 5;                                  \
  const int m0 = (wid >> 1) * 32;                                             \
  const int n0 = (wid & 1) * 64;                                              \
  const int r = lane & 15, h = lane >> 4;                                     \
  const int sr = tid >> 1, sc = (tid & 1) * 32;

// ---------------- GEMM1: H = silu(Xg @ W1[e]) in bf16 ----------------
__global__ __launch_bounds__(256) void gemm1_kernel(
    const unsigned short* __restrict__ xb, const unsigned short* __restrict__ W1t,
    const int* __restrict__ cnt, const int* __restrict__ baseo,
    const int* __restrict__ rowmap, unsigned short* __restrict__ Hraw) {
  const int e  = blockIdx.x & 7;
  const int mt = blockIdx.x >> 3;
  const int rows = cnt[e];
  if (mt * BM >= rows) return;
  const int f0 = blockIdx.y * BN;
  const int segbase = baseo[e];

  __shared__ __align__(16) unsigned short As[2][BM][PA];
  __shared__ __align__(16) unsigned short Bs[2][BN][PA];

  GEMM_PREAMBLE()

  const int ga = mt * BM + sr;
  const int tok = (ga < rows) ? rowmap[segbase + ga] : rowmap[segbase];
  const unsigned short* aptr = xb + (size_t)tok * Dm + sc;
  const unsigned short* bptr = W1t + ((size_t)e * Fm + f0 + sr) * Dm + sc;

  v8f acc[2][4];
#pragma unroll
  for (int am = 0; am < 2; ++am)
#pragma unroll
    for (int bn = 0; bn < 4; ++bn)
      acc[am][bn] = (v8f){0.f,0.f,0.f,0.f,0.f,0.f,0.f,0.f};

  auto compute = [&](int bsel, int ko) {
    V16U a[2], b[4];
#pragma unroll
    for (int am = 0; am < 2; ++am) {
      const unsigned short* ap = &As[bsel][m0 + am * 16 + r][ko];
      a[am].h[0] = *(const v8bf*)(ap + h * 8);
      a[am].h[1] = *(const v8bf*)(ap + 16 + h * 8);
    }
#pragma unroll
    for (int bn = 0; bn < 4; ++bn) {
      const unsigned short* bp = &Bs[bsel][n0 + bn * 16 + r][ko];
      b[bn].h[0] = *(const v8bf*)(bp + h * 8);
      b[bn].h[1] = *(const v8bf*)(bp + 16 + h * 8);
    }
#pragma unroll
    for (int am = 0; am < 2; ++am)
#pragma unroll
      for (int bn = 0; bn < 4; ++bn)
        acc[am][bn] = __builtin_amdgcn_wmma_f32_16x16x32_bf16(
            false, a[am].v, false, b[bn].v, (short)0, acc[am][bn], false, false);
  };

#if USE_ASYNC_LDS
  auto stage = [&](int b, int k0) {
#pragma unroll
    for (int q = 0; q < 4; ++q) {
      async_cp16(aptr + k0 + q * 8, &As[b][sr][sc + q * 8]);
      async_cp16(bptr + k0 + q * 8, &Bs[b][sr][sc + q * 8]);
    }
  };
  stage(0, 0);
  int buf = 0;
#pragma unroll 1
  for (int k0 = KC; k0 < Dm; k0 += KC) {
    wait_async0();
    __syncthreads();
    stage(buf ^ 1, k0);
    compute(buf, 0);
    compute(buf, 32);
    buf ^= 1;
  }
  wait_async0();
  __syncthreads();
  compute(buf, 0);
  compute(buf, 32);
#else
  {
#pragma unroll
    for (int q = 0; q < 4; ++q) {
      *(v8us*)&As[0][sr][sc + q * 8] = *(const v8us*)(aptr + q * 8);
      *(v8us*)&Bs[0][sr][sc + q * 8] = *(const v8us*)(bptr + q * 8);
    }
  }
  int buf = 0;
#pragma unroll 1
  for (int k0 = KC; k0 < Dm; k0 += KC) {
    v8us ra[4], rb[4];
#pragma unroll
    for (int q = 0; q < 4; ++q) {
      ra[q] = *(const v8us*)(aptr + k0 + q * 8);
      rb[q] = *(const v8us*)(bptr + k0 + q * 8);
    }
    __syncthreads();
    compute(buf, 0);
    compute(buf, 32);
#pragma unroll
    for (int q = 0; q < 4; ++q) {
      *(v8us*)&As[buf ^ 1][sr][sc + q * 8] = ra[q];
      *(v8us*)&Bs[buf ^ 1][sr][sc + q * 8] = rb[q];
    }
    buf ^= 1;
  }
  __syncthreads();
  compute(buf, 0);
  compute(buf, 32);
#endif

#pragma unroll
  for (int am = 0; am < 2; ++am)
#pragma unroll
    for (int i = 0; i < 8; ++i) {
      const int rowl = m0 + am * 16 + h * 8 + i;
      const int g = mt * BM + rowl;
      if (g < rows) {
        unsigned short* hp = Hraw + (size_t)(segbase + g) * Fm + f0 + n0;
        hp[r]      = f2bfu(siluf(acc[am][0][i]));
        hp[16 + r] = f2bfu(siluf(acc[am][1][i]));
        hp[32 + r] = f2bfu(siluf(acc[am][2][i]));
        hp[48 + r] = f2bfu(siluf(acc[am][3][i]));
      }
    }
}

// ---------------- GEMM2: y += w * (H @ W2[e]) ----------------
__global__ __launch_bounds__(256) void gemm2_kernel(
    const unsigned short* __restrict__ Hraw, const unsigned short* __restrict__ W2t,
    const int* __restrict__ cnt, const int* __restrict__ baseo,
    const int* __restrict__ rowmap, const float* __restrict__ wrow,
    float* __restrict__ y) {
  const int e  = blockIdx.x & 7;
  const int mt = blockIdx.x >> 3;
  const int rows = cnt[e];
  if (mt * BM >= rows) return;
  const int d0 = blockIdx.y * BN;
  const int segbase = baseo[e];

  __shared__ __align__(16) unsigned short As[2][BM][PA];
  __shared__ __align__(16) unsigned short Bs[2][BN][PA];

  GEMM_PREAMBLE()

  const int ga = mt * BM + sr;
  const int hrow = (ga < rows) ? (segbase + ga) : segbase;
  const unsigned short* aptr = Hraw + (size_t)hrow * Fm + sc;
  const unsigned short* bptr = W2t + ((size_t)e * Dm + d0 + sr) * Fm + sc;

  v8f acc[2][4];
#pragma unroll
  for (int am = 0; am < 2; ++am)
#pragma unroll
    for (int bn = 0; bn < 4; ++bn)
      acc[am][bn] = (v8f){0.f,0.f,0.f,0.f,0.f,0.f,0.f,0.f};

  auto compute = [&](int bsel, int ko) {
    V16U a[2], b[4];
#pragma unroll
    for (int am = 0; am < 2; ++am) {
      const unsigned short* ap = &As[bsel][m0 + am * 16 + r][ko];
      a[am].h[0] = *(const v8bf*)(ap + h * 8);
      a[am].h[1] = *(const v8bf*)(ap + 16 + h * 8);
    }
#pragma unroll
    for (int bn = 0; bn < 4; ++bn) {
      const unsigned short* bp = &Bs[bsel][n0 + bn * 16 + r][ko];
      b[bn].h[0] = *(const v8bf*)(bp + h * 8);
      b[bn].h[1] = *(const v8bf*)(bp + 16 + h * 8);
    }
#pragma unroll
    for (int am = 0; am < 2; ++am)
#pragma unroll
      for (int bn = 0; bn < 4; ++bn)
        acc[am][bn] = __builtin_amdgcn_wmma_f32_16x16x32_bf16(
            false, a[am].v, false, b[bn].v, (short)0, acc[am][bn], false, false);
  };

#if USE_ASYNC_LDS
  auto stage = [&](int b, int k0) {
#pragma unroll
    for (int q = 0; q < 4; ++q) {
      async_cp16(aptr + k0 + q * 8, &As[b][sr][sc + q * 8]);
      async_cp16(bptr + k0 + q * 8, &Bs[b][sr][sc + q * 8]);
    }
  };
  stage(0, 0);
  int buf = 0;
#pragma unroll 1
  for (int k0 = KC; k0 < Fm; k0 += KC) {
    wait_async0();
    __syncthreads();
    stage(buf ^ 1, k0);
    compute(buf, 0);
    compute(buf, 32);
    buf ^= 1;
  }
  wait_async0();
  __syncthreads();
  compute(buf, 0);
  compute(buf, 32);
#else
  {
#pragma unroll
    for (int q = 0; q < 4; ++q) {
      *(v8us*)&As[0][sr][sc + q * 8] = *(const v8us*)(aptr + q * 8);
      *(v8us*)&Bs[0][sr][sc + q * 8] = *(const v8us*)(bptr + q * 8);
    }
  }
  int buf = 0;
#pragma unroll 1
  for (int k0 = KC; k0 < Fm; k0 += KC) {
    v8us ra[4], rb[4];
#pragma unroll
    for (int q = 0; q < 4; ++q) {
      ra[q] = *(const v8us*)(aptr + k0 + q * 8);
      rb[q] = *(const v8us*)(bptr + k0 + q * 8);
    }
    __syncthreads();
    compute(buf, 0);
    compute(buf, 32);
#pragma unroll
    for (int q = 0; q < 4; ++q) {
      *(v8us*)&As[buf ^ 1][sr][sc + q * 8] = ra[q];
      *(v8us*)&Bs[buf ^ 1][sr][sc + q * 8] = rb[q];
    }
    buf ^= 1;
  }
  __syncthreads();
  compute(buf, 0);
  compute(buf, 32);
#endif

#pragma unroll
  for (int am = 0; am < 2; ++am)
#pragma unroll
    for (int i = 0; i < 8; ++i) {
      const int rowl = m0 + am * 16 + h * 8 + i;
      const int g = mt * BM + rowl;
      if (g < rows) {
        const int tok = rowmap[segbase + g];
        const float w = wrow[segbase + g];
        float* yp = y + (size_t)tok * Dm + d0 + n0;
        unsafeAtomicAdd(yp + r,      acc[am][0][i] * w);
        unsafeAtomicAdd(yp + 16 + r, acc[am][1][i] * w);
        unsafeAtomicAdd(yp + 32 + r, acc[am][2][i] * w);
        unsafeAtomicAdd(yp + 48 + r, acc[am][3][i] * w);
      }
    }
}

extern "C" void kernel_launch(void* const* d_in, const int* in_sizes, int n_in,
                              void* d_out, int out_size, void* d_ws, size_t ws_size,
                              hipStream_t stream) {
  const float* x  = (const float*)d_in[0];
  const float* Wg = (const float*)d_in[1];
  const float* W1 = (const float*)d_in[2];
  const float* W2 = (const float*)d_in[3];
  float* y = (float*)d_out;

  // workspace layout (~236 MB total)
  char* w = (char*)d_ws;
  int*   cnt    = (int*)(w + 0);
  int*   curs   = (int*)(w + 64);
  int*   baseo  = (int*)(w + 128);
  float* ssum   = (float*)(w + 256);
  int*   te     = (int*)(w + 4096);
  float* tw     = (float*)(w + 4096 + (size_t)NKa * 4);
  int*   rowmap = (int*)(w + 4096 + (size_t)NKa * 8);
  float* wrow   = (float*)(w + 4096 + (size_t)NKa * 12);
  const size_t WTB = (size_t)Em * Dm * Fm * 2;        // 33.55 MB
  unsigned short* xb  = (unsigned short*)(w + 4096 + (size_t)NKa * 16);
  unsigned short* W1t = (unsigned short*)((char*)xb + (size_t)Ntok * Dm * 2);
  unsigned short* W2t = (unsigned short*)((char*)W1t + WTB);
  unsigned short* H   = (unsigned short*)((char*)W2t + WTB);

  zero_meta_kernel<<<1, 64, 0, stream>>>(cnt, ssum);
  zero_y_kernel<<<(Ntok * (Dm / 4)) / 256, 256, 0, stream>>>(y);
  cvt_x_kernel<<<(Ntok * Dm / 8) / 256, 256, 0, stream>>>(x, xb);
  transpose_bf16_kernel<<<dim3(Fm / 64, Dm / 32, Em), 256, 0, stream>>>(W1, W1t, Dm, Fm);
  transpose_bf16_kernel<<<dim3(Dm / 64, Fm / 32, Em), 256, 0, stream>>>(W2, W2t, Fm, Dm);
  route_kernel<<<Ntok / 8, 256, 0, stream>>>(x, Wg, cnt, ssum, te, tw);
  prefix_aux_kernel<<<1, 1, 0, stream>>>(cnt, ssum, baseo, curs, y);
  scatter_kernel<<<Ntok / 256, 256, 0, stream>>>(te, tw, curs, rowmap, wrow);
  gemm1_kernel<<<dim3(Em * (Ntok / BM), Fm / BN), 256, 0, stream>>>(xb, W1t, cnt, baseo, rowmap, H);
  gemm2_kernel<<<dim3(Em * (Ntok / BM), Dm / BN), 256, 0, stream>>>(H, W2t, cnt, baseo, rowmap, wrow, y);
}
